// TMSS_75445395521988
// MI455X (gfx1250) — matrix operand order repaired and checked by reading, not compiled, not run on hardware
//
#include <hip/hip_runtime.h>
#include <hip/hip_bf16.h>
#include <math.h>
#include <stdint.h>

// ---------------------------------------------------------------- types
typedef __bf16 bf16;
typedef __attribute__((ext_vector_type(16))) __bf16 v16bf;
typedef __attribute__((ext_vector_type(8)))  __bf16 v8bf;
typedef __attribute__((ext_vector_type(4)))  __bf16 v4bf;
typedef __attribute__((ext_vector_type(8)))  float  v8f;
typedef unsigned int u32x4 __attribute__((ext_vector_type(4)));
typedef int          i32x4 __attribute__((ext_vector_type(4)));
typedef int          i32x8 __attribute__((ext_vector_type(8)));

union FragBF { v16bf v; v8bf h[2]; };

__device__ __forceinline__ v8f wmma_bf16(v16bf a, v16bf b, v8f c) {
  // D = A(16x32 bf16) * B(32x16 bf16) + C(16x16 f32)
  return __builtin_amdgcn_wmma_f32_16x16x32_bf16(
      /*neg_a=*/false, a, /*neg_b=*/false, b,
      /*c_mod=*/(short)0, c, /*reuse_a=*/false, /*reuse_b=*/false);
}

// model dims
#define BB     8
#define SS     96
#define NPATCH 216     // 6^3
#define NTOK   217
#define NPAD   224     // padded token count (14*16)
#define HIDD   768
#define QKVN   2304
#define HEADS  12
#define DHD    64
#define KCONV  33
#define PEK    16384   // 4*16*16*16
#define MROWS  (BB*NPAD)   // 1792
#define PEM    (BB*NPATCH) // 1728

// LDS A-tile row stride (bf16 elems): 64 data + 8 pad (matches TDM pad descriptor)
#define ASTR 72
#define TDM_ROWS 32
#define TDM_COLS 64

// ---------------------------------------------------------------- TDM plumbing
#if defined(__has_builtin)
#if __has_builtin(__builtin_amdgcn_tensor_load_to_lds)
#define HAVE_TDM 1
#if __has_include(<hip/amd_detail/amd_gfx1250_TDM.h>)
// amdgpu-toolchain (therock): 6-arg form
#define TDM_LOAD(g0, g1, g2, g3) \
  __builtin_amdgcn_tensor_load_to_lds(g0, g1, g2, g3, (i32x8){0,0,0,0,0,0,0,0}, 0)
#else
// ROCm 7.2 (clang-22): 5-arg form
#define TDM_LOAD(g0, g1, g2, g3) \
  __builtin_amdgcn_tensor_load_to_lds(g0, g1, g2, g3, 0)
#endif
#endif
#endif
#ifndef HAVE_TDM
#define HAVE_TDM 0
#endif

#if HAVE_TDM
// Issue a TDM load of a [32 rows][64 cols] bf16 tile (row stride K elems) from
// global into LDS at lds_off, padding each 128B row with 16B (-> ASTR=72 rows).
__device__ __forceinline__ void tdm_issue_a(unsigned lds_off, const bf16* gptr, int K) {
  const unsigned long long ga = (unsigned long long)(uintptr_t)gptr;
  u32x4 g0;
  g0.x = 1u;                                                  // count=1 (valid user D#)
  g0.y = lds_off;                                             // lds_addr (bytes)
  g0.z = (unsigned)(ga & 0xffffffffu);                        // global_addr[31:0]
  g0.w = (unsigned)((ga >> 32) & 0x01ffffffu) | (2u << 30);   // addr[56:32] | type=2
  i32x8 g1;
  g1[0] = (int)((1u << 16)      // data_size = 1 -> 2 bytes
              | (1u << 20)      // pad_enable
              | (4u << 22)      // pad_interval: 32 DWORDs (= one 128B row)
              | (3u << 25));    // pad_amount: 4 DWORDs (= 16B -> stride 144B)
  const unsigned td0 = TDM_COLS, td1 = TDM_ROWS;
  g1[1] = (int)((td0 & 0xffffu) << 16);                       // tensor_dim0 lo16
  g1[2] = (int)((td0 >> 16) | ((td1 & 0xffffu) << 16));       // dim0 hi / dim1 lo
  g1[3] = (int)((td1 >> 16) | ((unsigned)TDM_COLS << 16));    // dim1 hi / tile_dim0
  g1[4] = (int)TDM_ROWS;                                      // tile_dim1 (tile_dim2=0)
  g1[5] = (int)(unsigned)K;                                   // tensor_dim0_stride lo32
  g1[6] = 0;
  g1[7] = 0;
  const i32x4 z4 = {0, 0, 0, 0};
  TDM_LOAD(g0, g1, z4, z4);
}
#endif

// ---------------------------------------------------------------- weight prep
__global__ __launch_bounds__(256)
void k_convert_bf16(const float* __restrict__ in, bf16* __restrict__ out, size_t n) {
  size_t i = (size_t)blockIdx.x * 256 + threadIdx.x;
  if (i < n) out[i] = (bf16)in[i];
}

// out[l][n][k] = in[l][k][n]   (f32 [batch][K][N] -> bf16 [batch][N][K])
__global__ __launch_bounds__(256)
void k_transpose_bf16(const float* __restrict__ in, bf16* __restrict__ out,
                      int K, int N, size_t total) {
  size_t i = (size_t)blockIdx.x * 256 + threadIdx.x;
  if (i >= total) return;
  size_t kn = (size_t)K * N;
  size_t l = i / kn, rem = i % kn;
  size_t n = rem / K, k = rem % K;
  out[i] = (bf16)in[l * kn + k * N + n];
}

// ---------------------------------------------------------------- generic WMMA GEMM
// A: bf16 [M][K] row-major, Bt: bf16 [N][K], K % 64 == 0, M % 32 == 0, N % 256 == 0.
// Block = 256 threads = 8 waves; block tile 32(M) x 256(N); wave tile 16 x 64
// (4 independent accumulator chains). A tile staged to LDS via TDM (double
// buffered) when available, else cooperative ds stores.
// MODE 0: Cf = acc + bias            (f32 out)
// MODE 1: Cb = gelu(acc + bias)      (bf16 out, exact erf gelu)
// MODE 2: Cf += acc + bias           (residual accumulate)
template <int MODE>
__global__ __launch_bounds__(256)
void k_gemm(const bf16* __restrict__ A, const bf16* __restrict__ Bt,
            const float* __restrict__ bias,
            float* __restrict__ Cf, bf16* __restrict__ Cb,
            int K, int ldc) {
  __shared__ bf16 lds_a[2][TDM_ROWS * ASTR];
  const int t    = threadIdx.x;
  const int wave = t >> 5;
  const int lane = t & 31;
  const int hi   = lane >> 4;
  const int lm   = lane & 15;
  const int wm   = wave >> 2;  // 0..1 : M half
  const int wn   = wave & 3;   // 0..3 : N quarter
  const int m0   = blockIdx.y * 32;
  const int n0   = blockIdx.x * 256 + wn * 64;

  v8f acc[4];
#pragma unroll
  for (int nt = 0; nt < 4; ++nt) acc[nt] = (v8f){0.f,0.f,0.f,0.f,0.f,0.f,0.f,0.f};

  const bf16* __restrict__ bp = Bt + (size_t)(n0 + lm) * K;

#if HAVE_TDM
  const unsigned lds_off0 = (unsigned)(uintptr_t)&lds_a[0][0];
  const unsigned lds_off1 = (unsigned)(uintptr_t)&lds_a[1][0];
  const bf16* __restrict__ agbase = A + (size_t)m0 * K;
  if (wave == 0) tdm_issue_a(lds_off0, agbase, K);
#else
  const int arow = t >> 3;        // 0..31
  const int col8 = (t & 7) * 8;   // 0..56
  const bf16* __restrict__ aptr = A + (size_t)(m0 + arow) * K + col8;
#endif

  for (int kb = 0; kb < K; kb += 64) {
#if HAVE_TDM
    const int cur = (kb >> 6) & 1;
    if (wave == 0) __builtin_amdgcn_s_wait_tensorcnt((short)0);
    __syncthreads();
    if (wave == 0 && (kb + 64) < K)
      tdm_issue_a(cur ? lds_off0 : lds_off1, agbase + kb + 64, K);
    const bf16* __restrict__ lbase = &lds_a[cur][0];
#else
    *(v8bf*)&lds_a[0][arow * ASTR + col8] = *(const v8bf*)(aptr + kb);
    __syncthreads();
    const bf16* __restrict__ lbase = &lds_a[0][0];
#endif

#pragma unroll
    for (int kk = 0; kk < 2; ++kk) {
      FragBF fa;
      const bf16* ar = lbase + (wm * 16 + lm) * ASTR + kk * 32 + hi * 8;
      fa.h[0] = *(const v8bf*)(ar);
      fa.h[1] = *(const v8bf*)(ar + 16);
#pragma unroll
      for (int nt = 0; nt < 4; ++nt) {
        FragBF fb;
        const bf16* br = bp + (size_t)nt * 16 * K + kb + kk * 32 + hi * 8;
        fb.h[0] = *(const v8bf*)(br);
        fb.h[1] = *(const v8bf*)(br + 16);
        acc[nt] = wmma_bf16(fa.v, fb.v, acc[nt]);
      }
    }
#if !HAVE_TDM
    __syncthreads();
#endif
  }

  const int mrow0 = m0 + wm * 16;
#pragma unroll
  for (int nt = 0; nt < 4; ++nt) {
    const int col = n0 + nt * 16 + lm;
    const float bv = bias ? bias[col] : 0.f;
#pragma unroll
    for (int r = 0; r < 8; ++r) {
      const int row = mrow0 + r + hi * 8;
      const float val = acc[nt][r] + bv;
      const size_t idx = (size_t)row * ldc + col;
      if (MODE == 0) {
        Cf[idx] = val;
      } else if (MODE == 1) {
        Cb[idx] = (bf16)(0.5f * val * (1.f + erff(val * 0.70710678118654752f)));
      } else {
        Cf[idx] += val;
      }
    }
  }
}

// ---------------------------------------------------------------- patch-embed implicit GEMM
// M = 1728 (b,patch), N = 768, K = 16384. im2col of img staged into LDS as bf16.
// Block tile 32 x 256, wave tile 16 x 64 (same MMA core as k_gemm).
// Wt = conv_w as bf16 [768][16384] (already N-major). bias = conv_b.
__global__ __launch_bounds__(256)
void k_patch_gemm(const float* __restrict__ img, const bf16* __restrict__ Wt,
                  const float* __restrict__ bias, float* __restrict__ C) {
  __shared__ bf16 lds_a[TDM_ROWS * ASTR];
  const int t    = threadIdx.x;
  const int wave = t >> 5;
  const int lane = t & 31;
  const int hi   = lane >> 4;
  const int lm   = lane & 15;
  const int wm   = wave >> 2;
  const int wn   = wave & 3;
  const int m0   = blockIdx.y * 32;
  const int n0   = blockIdx.x * 256 + wn * 64;

  v8f acc[4];
#pragma unroll
  for (int nt = 0; nt < 4; ++nt) acc[nt] = (v8f){0.f,0.f,0.f,0.f,0.f,0.f,0.f,0.f};

  // im2col source for this thread's staged 8 elems
  const int arow = t >> 3;        // 0..31 : patch row in tile
  const int col8 = (t & 7) * 8;   // 0..56 : k offset in 64-chunk
  const int r  = m0 + arow;
  const int b  = r / NPATCH;
  const int p  = r % NPATCH;
  const int pz = p / 36, py = (p / 6) % 6, px = p % 6;
  const float* __restrict__ base =
      img + (size_t)b * 4 * SS * SS * SS + (size_t)(py * 16) * SS + (px * 16);
  const bf16* __restrict__ bp = Wt + (size_t)(n0 + lm) * PEK;

  for (int kb = 0; kb < PEK; kb += 64) {
    const int k  = kb + col8;
    const int c  = k >> 12;
    const int dz = (k >> 8) & 15;
    const int dy = (k >> 4) & 15;
    const int dx = k & 15;  // 0 or 8
    const float* src = &base[((size_t)c * SS + (size_t)(pz * 16 + dz)) * (size_t)(SS * SS)
                             + (size_t)dy * SS + dx];
    const float4 f0 = *(const float4*)(src);
    const float4 f1 = *(const float4*)(src + 4);
    v8bf q = {(bf16)f0.x, (bf16)f0.y, (bf16)f0.z, (bf16)f0.w,
              (bf16)f1.x, (bf16)f1.y, (bf16)f1.z, (bf16)f1.w};
    *(v8bf*)&lds_a[arow * ASTR + col8] = q;
    __syncthreads();

#pragma unroll
    for (int kk = 0; kk < 2; ++kk) {
      FragBF fa;
      const bf16* ar = &lds_a[(wm * 16 + lm) * ASTR + kk * 32 + hi * 8];
      fa.h[0] = *(const v8bf*)(ar);
      fa.h[1] = *(const v8bf*)(ar + 16);
#pragma unroll
      for (int nt = 0; nt < 4; ++nt) {
        FragBF fb;
        const bf16* br = bp + (size_t)nt * 16 * PEK + kb + kk * 32 + hi * 8;
        fb.h[0] = *(const v8bf*)(br);
        fb.h[1] = *(const v8bf*)(br + 16);
        acc[nt] = wmma_bf16(fa.v, fb.v, acc[nt]);
      }
    }
    __syncthreads();
  }

  const int mrow0 = m0 + wm * 16;
#pragma unroll
  for (int nt = 0; nt < 4; ++nt) {
    const int col = n0 + nt * 16 + lm;
    const float bv = bias[col];
#pragma unroll
    for (int rr = 0; rr < 8; ++rr) {
      const int row = mrow0 + rr + hi * 8;
      C[(size_t)row * HIDD + col] = acc[nt][rr] + bv;
    }
  }
}

// ---------------------------------------------------------------- X assembly
// X f32 [8][224][768]: row 0 = clin@ehr_w + ehr_b + pos, rows 1..216 = PE + pos, rows>=217 = 0
__global__ __launch_bounds__(256)
void k_assemble_x(const float* __restrict__ PE, const float* __restrict__ clin,
                  const float* __restrict__ ehr_w, const float* __restrict__ ehr_b,
                  const float* __restrict__ pos_emb, float* __restrict__ X) {
  size_t i = (size_t)blockIdx.x * 256 + threadIdx.x;
  if (i >= (size_t)BB * NPAD * HIDD) return;
  const int c = (int)(i % HIDD);
  const int n = (int)((i / HIDD) % NPAD);
  const int b = (int)(i / ((size_t)NPAD * HIDD));
  float v;
  if (n == 0) {
    float s = ehr_b[c];
#pragma unroll
    for (int k = 0; k < 16; ++k) s += clin[b * 16 + k] * ehr_w[k * HIDD + c];
    v = s + pos_emb[c];
  } else if (n <= NPATCH) {
    v = PE[((size_t)b * NPATCH + (n - 1)) * HIDD + c] + pos_emb[(size_t)n * HIDD + c];
  } else {
    v = 0.f;
  }
  X[i] = v;
}

// ---------------------------------------------------------------- LayerNorm -> bf16
__global__ __launch_bounds__(256)
void k_layernorm(const float* __restrict__ X, const float* __restrict__ w,
                 const float* __restrict__ bsh, bf16* __restrict__ out) {
  const int row = blockIdx.x;
  const int t = threadIdx.x;
  const float* __restrict__ x = X + (size_t)row * HIDD;
  __shared__ float red[256];
  const float v0 = x[t], v1 = x[t + 256], v2 = x[t + 512];
  red[t] = v0 + v1 + v2;
  __syncthreads();
  for (int o = 128; o; o >>= 1) { if (t < o) red[t] += red[t + o]; __syncthreads(); }
  const float mean = red[0] * (1.f / HIDD);
  __syncthreads();
  red[t] = v0 * v0 + v1 * v1 + v2 * v2;
  __syncthreads();
  for (int o = 128; o; o >>= 1) { if (t < o) red[t] += red[t + o]; __syncthreads(); }
  const float var = red[0] * (1.f / HIDD) - mean * mean;
  const float rs = rsqrtf(var + 1e-5f);
  bf16* __restrict__ o = out + (size_t)row * HIDD;
  o[t]       = (bf16)((v0 - mean) * rs * w[t]       + bsh[t]);
  o[t + 256] = (bf16)((v1 - mean) * rs * w[t + 256] + bsh[t + 256]);
  o[t + 512] = (bf16)((v2 - mean) * rs * w[t + 512] + bsh[t + 512]);
}

// ---------------------------------------------------------------- attention + depthwise conv + residual
// softmax over the single patch-token key is identity -> rows n>=1 attend only to v[0].
// Row 0 does the full 217-way softmax. Plus per-head 33-tap conv over tokens. X += result.
__global__ __launch_bounds__(256)
void k_attention(const float* __restrict__ QKV, const float* __restrict__ res_w_l,
                 float* __restrict__ X) {
  const int bh = blockIdx.x;
  const int b = bh / HEADS, h = bh % HEADS;
  const int t = threadIdx.x;
  __shared__ float vs[NTOK * DHD];   // V for this (b,h): 217x64
  __shared__ float q0[DHD];
  __shared__ float wk[KCONV];
  __shared__ float lg[NPAD];
  __shared__ float red[256];
  __shared__ float out0[DHD];

  const float* __restrict__ qkvb = QKV + (size_t)b * NPAD * QKVN;
  for (int i = t; i < NTOK * DHD; i += 256) {
    const int n = i >> 6, d = i & 63;
    vs[i] = qkvb[(size_t)n * QKVN + 2 * HIDD + h * DHD + d];
  }
  if (t < DHD) q0[t] = qkvb[h * DHD + t] * 0.125f;  // DH^-0.5
  if (t < KCONV) wk[t] = res_w_l[h * KCONV + t];
  __syncthreads();

  float l = -3.0e38f;
  if (t < NTOK) {
    const float* __restrict__ kr = qkvb + (size_t)t * QKVN + HIDD + h * DHD;
    float s = 0.f;
#pragma unroll 8
    for (int d = 0; d < DHD; ++d) s += q0[d] * kr[d];
    l = s;
  }
  red[t] = l;
  __syncthreads();
  for (int o = 128; o; o >>= 1) { if (t < o) red[t] = fmaxf(red[t], red[t + o]); __syncthreads(); }
  const float mx = red[0];
  __syncthreads();
  const float e = (t < NTOK) ? __expf(l - mx) : 0.f;
  red[t] = e;
  __syncthreads();
  for (int o = 128; o; o >>= 1) { if (t < o) red[t] += red[t + o]; __syncthreads(); }
  const float inv = 1.f / red[0];
  __syncthreads();
  if (t < NPAD) lg[t] = (t < NTOK) ? e * inv : 0.f;
  __syncthreads();

  if (t < DHD) {
    float s = 0.f;
    for (int j = 0; j < NTOK; ++j) s += lg[j] * vs[j * DHD + t];
    out0[t] = s;
  }
  __syncthreads();

  float* __restrict__ xb = X + (size_t)b * NPAD * HIDD + h * DHD;
  for (int i = t; i < NTOK * DHD; i += 256) {
    const int n = i >> 6, d = i & 63;
    float r = 0.f;
    for (int k2 = 0; k2 < KCONV; ++k2) {
      const int j = n + k2 - (KCONV / 2);
      if (j >= 0 && j < NTOK) r += vs[j * DHD + d] * wk[k2];
    }
    const float basev = (n == 0) ? out0[d] : vs[d];  // vs[d] == v[token0][d]
    xb[(size_t)n * HIDD + d] += basev + r;
  }
}

// ---------------------------------------------------------------- final strided copy
__global__ __launch_bounds__(256)
void k_final_copy(const float* __restrict__ X, float* __restrict__ out) {
  size_t i = (size_t)blockIdx.x * 256 + threadIdx.x;
  if (i >= (size_t)BB * NTOK * HIDD) return;
  const int c = (int)(i % HIDD);
  const int n = (int)((i / HIDD) % NTOK);
  const int b = (int)(i / ((size_t)NTOK * HIDD));
  out[i] = X[((size_t)b * NPAD + n) * HIDD + c];
}

// ---------------------------------------------------------------- launcher
extern "C" void kernel_launch(void* const* d_in, const int* in_sizes, int n_in,
                              void* d_out, int out_size, void* d_ws, size_t ws_size,
                              hipStream_t stream) {
  (void)in_sizes; (void)n_in; (void)out_size; (void)ws_size;
  const float* img     = (const float*)d_in[0];
  const float* clin    = (const float*)d_in[1];
  const float* conv_w  = (const float*)d_in[2];
  const float* conv_b  = (const float*)d_in[3];
  const float* ehr_w   = (const float*)d_in[4];
  const float* ehr_b   = (const float*)d_in[5];
  const float* pos_emb = (const float*)d_in[6];
  const float* ln_w    = (const float*)d_in[7];
  const float* ln_b    = (const float*)d_in[8];
  const float* qkv_w   = (const float*)d_in[9];
  const float* res_w   = (const float*)d_in[10];
  const float* ff_ln_w = (const float*)d_in[11];
  const float* ff_ln_b = (const float*)d_in[12];
  const float* w1      = (const float*)d_in[13];
  const float* b1      = (const float*)d_in[14];
  const float* w2      = (const float*)d_in[15];
  const float* b2      = (const float*)d_in[16];
  float* out = (float*)d_out;

  // workspace carve-up (256B aligned)
  char* ws = (char*)d_ws;
  size_t off = 0;
  auto carve = [&](size_t bytes) { char* p = ws + off; off += (bytes + 255) & ~(size_t)255; return p; };
  bf16*  convw_bf = (bf16*)carve((size_t)HIDD * PEK * 2);           // [768][16384]
  bf16*  qkvT     = (bf16*)carve((size_t)4 * QKVN * HIDD * 2);      // [L][2304][768]
  bf16*  w1T      = (bf16*)carve((size_t)4 * HIDD * HIDD * 2);      // [L][768][768]
  bf16*  w2T      = (bf16*)carve((size_t)4 * HIDD * HIDD * 2);
  float* PE       = (float*)carve((size_t)PEM * HIDD * 4);          // [1728][768]
  float* X        = (float*)carve((size_t)MROWS * HIDD * 4);        // [1792][768]
  bf16*  XLN      = (bf16*)carve((size_t)MROWS * HIDD * 2);         // [1792][768]
  float* QKV      = (float*)carve((size_t)MROWS * QKVN * 4);        // [1792][2304]
  bf16*  FF1      = (bf16*)carve((size_t)MROWS * HIDD * 2);         // [1792][768]

  // 1) weight prep (recomputed each call: deterministic, capture-safe)
  {
    size_t n = (size_t)HIDD * PEK;
    k_convert_bf16<<<(unsigned)((n + 255) / 256), 256, 0, stream>>>(conv_w, convw_bf, n);
    size_t tq = (size_t)4 * HIDD * QKVN;
    k_transpose_bf16<<<(unsigned)((tq + 255) / 256), 256, 0, stream>>>(qkv_w, qkvT, HIDD, QKVN, tq);
    size_t tw = (size_t)4 * HIDD * HIDD;
    k_transpose_bf16<<<(unsigned)((tw + 255) / 256), 256, 0, stream>>>(w1, w1T, HIDD, HIDD, tw);
    k_transpose_bf16<<<(unsigned)((tw + 255) / 256), 256, 0, stream>>>(w2, w2T, HIDD, HIDD, tw);
  }

  // 2) patch embedding GEMM: PE = im2col(img) @ conv_w^T + conv_b
  k_patch_gemm<<<dim3(HIDD / 256, PEM / 32), 256, 0, stream>>>(img, convw_bf, conv_b, PE);

  // 3) assemble X (+EHR token, +pos_emb, zero padding rows)
  {
    size_t n = (size_t)BB * NPAD * HIDD;
    k_assemble_x<<<(unsigned)((n + 255) / 256), 256, 0, stream>>>(PE, clin, ehr_w, ehr_b, pos_emb, X);
  }

  // 4) transformer layers
  for (int l = 0; l < 4; ++l) {
    // attention branch
    k_layernorm<<<MROWS, 256, 0, stream>>>(X, ln_w + l * HIDD, ln_b + l * HIDD, XLN);
    k_gemm<0><<<dim3(QKVN / 256, MROWS / 32), 256, 0, stream>>>(
        XLN, qkvT + (size_t)l * QKVN * HIDD, nullptr, QKV, nullptr, HIDD, QKVN);
    k_attention<<<BB * HEADS, 256, 0, stream>>>(QKV, res_w + (size_t)l * HEADS * KCONV, X);
    // MLP branch
    k_layernorm<<<MROWS, 256, 0, stream>>>(X, ff_ln_w + l * HIDD, ff_ln_b + l * HIDD, XLN);
    k_gemm<1><<<dim3(HIDD / 256, MROWS / 32), 256, 0, stream>>>(
        XLN, w1T + (size_t)l * HIDD * HIDD, b1 + l * HIDD, nullptr, FF1, HIDD, HIDD);
    k_gemm<2><<<dim3(HIDD / 256, MROWS / 32), 256, 0, stream>>>(
        FF1, w2T + (size_t)l * HIDD * HIDD, b2 + l * HIDD, X, nullptr, HIDD, HIDD);
  }

  // 5) output copy (drop padding rows)
  {
    size_t n = (size_t)BB * NTOK * HIDD;
    k_final_copy<<<(unsigned)((n + 255) / 256), 256, 0, stream>>>(X, out);
  }
}
